// LSTMAttnDecoder_8504035246533
// MI455X (gfx1250) — matrix operand order repaired and checked by reading, not compiled, not run on hardware
//
#include <hip/hip_runtime.h>
#include <hip/hip_bf16.h>

// ---------------------------------------------------------------------------
// Types for CDNA5 WMMA (gfx1250): v_wmma_f32_16x16x32_bf16
// ---------------------------------------------------------------------------
typedef __bf16 bf16_t;
typedef __attribute__((ext_vector_type(16))) __bf16 v16bf;
typedef __attribute__((ext_vector_type(8)))  float  v8f;
typedef __attribute__((ext_vector_type(4)))  unsigned int uint4v;

union FragBF { v16bf v; uint4v u[2]; };

// A-matrix fragment (16x32 bf16, ISA 7.12.2): lanes 0-15 hold M=lane,
// K=0..7 in v0..v3 and K=16..23 in v4..v7; lanes 16-31 hold K=8..15 / 24..31.
__device__ __forceinline__ v16bf load_frag_a(const bf16_t* __restrict__ base,
                                             int lda, int row0, int k0, int lane) {
  int m    = lane & 15;
  int koff = (lane >> 4) << 3;                 // 0 or 8
  const bf16_t* p = base + (size_t)(row0 + m) * lda + k0 + koff;
  FragBF f;
  f.u[0] = *(const uint4v*)(p);                // K = koff .. koff+7
  f.u[1] = *(const uint4v*)(p + 16);           // K = 16+koff .. 16+koff+7
  return f.v;
}

// B-matrix fragment (32x16 bf16): lane n holds column N=n; lanes 0-15 carry
// K=0..7 / 16..23, lanes 16-31 carry K=8..15 / 24..31 (same as A).
__device__ __forceinline__ v16bf load_frag_b(const bf16_t* __restrict__ base,
                                             int ldb, int col0, int k0, int lane) {
  int n    = lane & 15;
  int koff = (lane >> 4) << 3;                 // 0 or 8
  const bf16_t* p = base + (size_t)(col0 + n) * ldb + k0 + koff;
  FragBF f;
  f.u[0] = *(const uint4v*)(p);                // K = koff .. koff+7
  f.u[1] = *(const uint4v*)(p + 16);           // K = 16+koff .. 16+koff+7
  return f.v;
}

// ---------------------------------------------------------------------------
// WMMA GEMM:  C[M,N] = A[M,K] * B[N,K]^T (+ bias) (+= if accumulate)
// One wave = 16x64 tile. 8 waves/block. Requires M%16==0, N%64==0, K%32==0.
// All control flow is wave-uniform & scalarized -> EXEC stays all-1 for wmma.
// K-loop is software-pipelined (prefetch next fragments before wmma issue).
// ---------------------------------------------------------------------------
__global__ __launch_bounds__(256)
void wmma_gemm_bf16(const bf16_t* __restrict__ A, int lda,
                    const bf16_t* __restrict__ B, int ldb,
                    float* __restrict__ C, long long ldc,
                    const float* __restrict__ bias,
                    int M, int N, int K, int accumulate) {
  const int lane = threadIdx.x & 31;
  const int wave = __builtin_amdgcn_readfirstlane(threadIdx.x >> 5);
  const int tilesM = M >> 4;
  const int tilesG = N >> 6;                   // groups of 4 16-wide tiles
  const int gidx   = blockIdx.x * 8 + wave;    // fully scalar
  if (gidx >= tilesM * tilesG) return;         // scalar branch, EXEC untouched
  const int tm   = gidx % tilesM;
  const int tg   = gidx / tilesM;
  const int row0 = tm << 4;
  const int col0 = tg << 6;

  const int rIT = (lane >> 4) << 3;            // C/D: VGPR i -> M = rIT + i
  const int cIL = lane & 15;                   // C/D: N = lane & 15

  v8f acc0 = {}, acc1 = {}, acc2 = {}, acc3 = {};
  if (accumulate) {
#pragma unroll
    for (int i = 0; i < 8; ++i) {
      const float* cr = C + (size_t)(row0 + rIT + i) * (size_t)ldc + cIL;
      acc0[i] = cr[col0];
      acc1[i] = cr[col0 + 16];
      acc2[i] = cr[col0 + 32];
      acc3[i] = cr[col0 + 48];
    }
  }

  // prologue: fragments for k=0
  v16bf fa  = load_frag_a(A, lda, row0, 0, lane);
  v16bf fb0 = load_frag_b(B, ldb, col0,      0, lane);
  v16bf fb1 = load_frag_b(B, ldb, col0 + 16, 0, lane);
  v16bf fb2 = load_frag_b(B, ldb, col0 + 32, 0, lane);
  v16bf fb3 = load_frag_b(B, ldb, col0 + 48, 0, lane);

  for (int k0 = 32; k0 < K; k0 += 32) {        // pipelined main loop
    v16bf na  = load_frag_a(A, lda, row0, k0, lane);
    v16bf nb0 = load_frag_b(B, ldb, col0,      k0, lane);
    v16bf nb1 = load_frag_b(B, ldb, col0 + 16, k0, lane);
    v16bf nb2 = load_frag_b(B, ldb, col0 + 32, k0, lane);
    v16bf nb3 = load_frag_b(B, ldb, col0 + 48, k0, lane);
    acc0 = __builtin_amdgcn_wmma_f32_16x16x32_bf16(false, fa, false, fb0, (short)0, acc0, false, false);
    acc1 = __builtin_amdgcn_wmma_f32_16x16x32_bf16(false, fa, false, fb1, (short)0, acc1, false, false);
    acc2 = __builtin_amdgcn_wmma_f32_16x16x32_bf16(false, fa, false, fb2, (short)0, acc2, false, false);
    acc3 = __builtin_amdgcn_wmma_f32_16x16x32_bf16(false, fa, false, fb3, (short)0, acc3, false, false);
    fa = na; fb0 = nb0; fb1 = nb1; fb2 = nb2; fb3 = nb3;
  }
  // epilogue wmmas
  acc0 = __builtin_amdgcn_wmma_f32_16x16x32_bf16(false, fa, false, fb0, (short)0, acc0, false, false);
  acc1 = __builtin_amdgcn_wmma_f32_16x16x32_bf16(false, fa, false, fb1, (short)0, acc1, false, false);
  acc2 = __builtin_amdgcn_wmma_f32_16x16x32_bf16(false, fa, false, fb2, (short)0, acc2, false, false);
  acc3 = __builtin_amdgcn_wmma_f32_16x16x32_bf16(false, fa, false, fb3, (short)0, acc3, false, false);

  float bv0 = 0.f, bv1 = 0.f, bv2 = 0.f, bv3 = 0.f;
  if (bias) {
    bv0 = bias[col0 + cIL];
    bv1 = bias[col0 + 16 + cIL];
    bv2 = bias[col0 + 32 + cIL];
    bv3 = bias[col0 + 48 + cIL];
  }
#pragma unroll
  for (int i = 0; i < 8; ++i) {
    float* cr = C + (size_t)(row0 + rIT + i) * (size_t)ldc + cIL;
    cr[col0]      = acc0[i] + bv0;
    cr[col0 + 16] = acc1[i] + bv1;
    cr[col0 + 32] = acc2[i] + bv2;
    cr[col0 + 48] = acc3[i] + bv3;
  }
}

// ---------------------------------------------------------------------------
// f32 -> bf16 conversion (plain, and zero-padded variant)
// ---------------------------------------------------------------------------
__global__ __launch_bounds__(256)
void cvt_f32_bf16(const float* __restrict__ src, bf16_t* __restrict__ dst, int n) {
  int i = blockIdx.x * 256 + threadIdx.x;
  if (i < n) dst[i] = (bf16_t)src[i];
}

__global__ __launch_bounds__(256)
void cvt_f32_bf16_pad(const float* __restrict__ src, bf16_t* __restrict__ dst,
                      int n_src, int n_total) {
  int i = blockIdx.x * 256 + threadIdx.x;
  if (i < n_total) dst[i] = (i < n_src) ? (bf16_t)src[i] : (bf16_t)0.f;
}

// x_seq_bf16[b,t,d] = bf16(emb[tgt_ids[b,t], d]);  B=32,T=128,D=512
__global__ __launch_bounds__(256)
void gather_embed(const float* __restrict__ emb, const int* __restrict__ ids,
                  bf16_t* __restrict__ x_seq, int total) {
  int i = blockIdx.x * 256 + threadIdx.x;
  if (i >= total) return;
  int d  = i & 511;
  int bt = i >> 9;                 // b*128 + t
  int id = ids[bt];
  x_seq[i] = (bf16_t)emb[(size_t)id * 512 + d];
}

// ---------------------------------------------------------------------------
// Attention scores: s[b,n] = sum_d tanh(hWh[b,d] + Wk_keys[b,n,d]) * attn_v[d]
// ---------------------------------------------------------------------------
__global__ __launch_bounds__(256)
void attn_scores(const float* __restrict__ hWh, const float* __restrict__ Wk,
                 const float* __restrict__ attn_v, float* __restrict__ scores) {
  int b = blockIdx.x;
  int wave = threadIdx.x >> 5, lane = threadIdx.x & 31;
  const float* hb = hWh + (size_t)b * 512;
  for (int n = wave; n < 196; n += 8) {
    const float* wk = Wk + ((size_t)b * 196 + n) * 512;
    float s = 0.f;
    for (int d = lane; d < 512; d += 32)
      s += tanhf(hb[d] + wk[d]) * attn_v[d];
#pragma unroll
    for (int off = 16; off > 0; off >>= 1)
      s += __shfl_xor(s, off, 32);
    if (lane == 0) scores[b * 196 + n] = s;
  }
}

// ---------------------------------------------------------------------------
// Softmax over N=196 + ctx[b,d] = sum_n alpha[n]*keys[b,n,d];
// writes z_bf16[b,0:512]=x_t and z_bf16[b,512:1024]=ctx (LSTM input).
// ---------------------------------------------------------------------------
__global__ __launch_bounds__(256)
void softmax_ctx(const float* __restrict__ scores, const float* __restrict__ keys,
                 const bf16_t* __restrict__ x_seq, int t, bf16_t* __restrict__ z) {
  __shared__ float alpha[196];
  __shared__ float red[256];
  int b = blockIdx.x, tid = threadIdx.x;
  float s = (tid < 196) ? scores[b * 196 + tid] : -3.4e38f;
  red[tid] = s; __syncthreads();
  for (int off = 128; off > 0; off >>= 1) {
    if (tid < off) red[tid] = fmaxf(red[tid], red[tid + off]);
    __syncthreads();
  }
  float mx = red[0]; __syncthreads();
  float e = (tid < 196) ? __expf(s - mx) : 0.f;
  red[tid] = e; __syncthreads();
  for (int off = 128; off > 0; off >>= 1) {
    if (tid < off) red[tid] += red[tid + off];
    __syncthreads();
  }
  float inv = 1.f / red[0];
  if (tid < 196) alpha[tid] = e * inv;
  __syncthreads();
  for (int d = tid; d < 512; d += 256) {
    float acc = 0.f;
    const float* kb = keys + (size_t)b * 196 * 512 + d;
    for (int n = 0; n < 196; ++n) acc += alpha[n] * kb[(size_t)n * 512];
    z[b * 1024 + 512 + d] = (bf16_t)acc;
    z[b * 1024 + d]       = x_seq[((size_t)b * 128 + t) * 512 + d];
  }
}

// ---------------------------------------------------------------------------
// LSTM cell (torch gate order i,f,g,o) + LayerNorm; writes h (f32+bf16), c.
// ---------------------------------------------------------------------------
__device__ __forceinline__ float sigm(float x) { return 1.f / (1.f + __expf(-x)); }

__global__ __launch_bounds__(256)
void lstm_cell_ln(const float* __restrict__ gates,
                  const float* __restrict__ b_ih, const float* __restrict__ b_hh,
                  float* __restrict__ c,
                  const float* __restrict__ ln_w, const float* __restrict__ ln_b,
                  float* __restrict__ h, bf16_t* __restrict__ h_bf) {
  __shared__ float red[256], red2[256];
  __shared__ float hr[512];
  int b = blockIdx.x, tid = threadIdx.x;
  const float* gb = gates + (size_t)b * 2048;
  float sum = 0.f, sumsq = 0.f;
#pragma unroll
  for (int j = 0; j < 2; ++j) {
    int d = tid + j * 256;
    float ig = gb[d]        + b_ih[d]        + b_hh[d];
    float fg = gb[512 + d]  + b_ih[512 + d]  + b_hh[512 + d];
    float gg = gb[1024 + d] + b_ih[1024 + d] + b_hh[1024 + d];
    float og = gb[1536 + d] + b_ih[1536 + d] + b_hh[1536 + d];
    float cn = sigm(fg) * c[b * 512 + d] + sigm(ig) * tanhf(gg);
    float hraw = sigm(og) * tanhf(cn);
    c[b * 512 + d] = cn;
    hr[d] = hraw;
    sum += hraw; sumsq += hraw * hraw;
  }
  red[tid] = sum; red2[tid] = sumsq; __syncthreads();
  for (int off = 128; off > 0; off >>= 1) {
    if (tid < off) { red[tid] += red[tid + off]; red2[tid] += red2[tid + off]; }
    __syncthreads();
  }
  float mu   = red[0] * (1.f / 512.f);
  float var  = red2[0] * (1.f / 512.f) - mu * mu;
  float rstd = rsqrtf(var + 1e-5f);
#pragma unroll
  for (int j = 0; j < 2; ++j) {
    int d = tid + j * 256;
    float hn = (hr[d] - mu) * rstd * ln_w[d] + ln_b[d];
    h[b * 512 + d]    = hn;
    h_bf[b * 512 + d] = (bf16_t)hn;
  }
}

// out[b, t, v] = logits_pad[b*10048 + v] + out_b[v];  grid (ceil(10000/256), 32)
__global__ __launch_bounds__(256)
void logits_bias_copy(const float* __restrict__ lp, const float* __restrict__ out_b,
                      float* __restrict__ out, int t) {
  int v = blockIdx.x * 256 + threadIdx.x;
  int b = blockIdx.y;
  if (v < 10000)
    out[((size_t)b * 128 + t) * 10000 + v] = lp[(size_t)b * 10048 + v] + out_b[v];
}

// ---------------------------------------------------------------------------
// Host side
// ---------------------------------------------------------------------------
static inline void launch_gemm(const bf16_t* A, int lda, const bf16_t* B, int ldb,
                               float* C, long long ldc, const float* bias,
                               int M, int N, int K, int acc, hipStream_t s) {
  int tilesM = M >> 4;
  int tilesG = N >> 6;                 // N must be a multiple of 64
  int total  = tilesM * tilesG;
  int blocks = (total + 7) >> 3;
  wmma_gemm_bf16<<<blocks, 256, 0, s>>>(A, lda, B, ldb, C, ldc, bias, M, N, K, acc);
}

static inline void launch_cvt(const float* src, bf16_t* dst, int n, hipStream_t s) {
  cvt_f32_bf16<<<(n + 255) / 256, 256, 0, s>>>(src, dst, n);
}

extern "C" void kernel_launch(void* const* d_in, const int* in_sizes, int n_in,
                              void* d_out, int out_size, void* d_ws, size_t ws_size,
                              hipStream_t stream) {
  (void)in_sizes; (void)n_in; (void)out_size; (void)ws_size;
  const int Bn = 32, Nn = 196, Tn = 128, DI = 768, DH = 512, Vn = 10000;
  const int Vp = 10048;                // V padded to multiple of 64

  const float* patches = (const float*)d_in[0];
  const float* cls     = (const float*)d_in[1];
  const int*   tgt_ids = (const int*)  d_in[2];
  const float* emb     = (const float*)d_in[3];
  const float* kv_W    = (const float*)d_in[4];
  const float* kv_b    = (const float*)d_in[5];
  const float* ih_W    = (const float*)d_in[6];
  const float* ih_b    = (const float*)d_in[7];
  const float* ic_W    = (const float*)d_in[8];
  const float* ic_b    = (const float*)d_in[9];
  const float* attn_Wh = (const float*)d_in[10];
  const float* attn_Wk = (const float*)d_in[11];
  const float* attn_v  = (const float*)d_in[12];
  const float* W_ih    = (const float*)d_in[13];
  const float* W_hh    = (const float*)d_in[14];
  const float* b_ih    = (const float*)d_in[15];
  const float* b_hh    = (const float*)d_in[16];
  const float* ln_w    = (const float*)d_in[17];
  const float* ln_b    = (const float*)d_in[18];
  const float* out_W   = (const float*)d_in[19];
  const float* out_b   = (const float*)d_in[20];
  float* out = (float*)d_out;

  // --- workspace arena (256B aligned) ---
  char* ws = (char*)d_ws;
  size_t off = 0;
  auto carve = [&](size_t bytes) -> char* {
    char* p = ws + off;
    off = (off + bytes + 255) & ~(size_t)255;
    return p;
  };
  bf16_t* pbf      = (bf16_t*)carve((size_t)Bn * Nn * DI * 2);   // patches bf16
  bf16_t* clsbf    = (bf16_t*)carve((size_t)Bn * DI * 2);
  bf16_t* kvWbf    = (bf16_t*)carve((size_t)DH * DI * 2);
  bf16_t* ihWbf    = (bf16_t*)carve((size_t)DH * DI * 2);
  bf16_t* icWbf    = (bf16_t*)carve((size_t)DH * DI * 2);
  bf16_t* aWhbf    = (bf16_t*)carve((size_t)DH * DH * 2);
  bf16_t* aWkbf    = (bf16_t*)carve((size_t)DH * DH * 2);
  bf16_t* Wihbf    = (bf16_t*)carve((size_t)4 * DH * 2 * DH * 2);
  bf16_t* Whhbf    = (bf16_t*)carve((size_t)4 * DH * DH * 2);
  bf16_t* outWbf   = (bf16_t*)carve((size_t)Vp * DH * 2);        // zero-padded
  bf16_t* xseqbf   = (bf16_t*)carve((size_t)Bn * Tn * DH * 2);
  float*  keys     = (float*) carve((size_t)Bn * Nn * DH * 4);
  bf16_t* keysbf   = (bf16_t*)carve((size_t)Bn * Nn * DH * 2);
  float*  Wkkeys   = (float*) carve((size_t)Bn * Nn * DH * 4);
  float*  h        = (float*) carve((size_t)Bn * DH * 4);
  bf16_t* hbf      = (bf16_t*)carve((size_t)Bn * DH * 2);
  float*  c        = (float*) carve((size_t)Bn * DH * 4);
  float*  hWh      = (float*) carve((size_t)Bn * DH * 4);
  float*  scores   = (float*) carve((size_t)Bn * Nn * 4);
  bf16_t* zbf      = (bf16_t*)carve((size_t)Bn * 2 * DH * 2);
  float*  gates    = (float*) carve((size_t)Bn * 4 * DH * 4);
  float*  lpad     = (float*) carve((size_t)Bn * Vp * 4);        // padded logits

  // --- one-time conversions ---
  launch_cvt(patches, pbf,   Bn * Nn * DI, stream);
  launch_cvt(cls,     clsbf, Bn * DI,      stream);
  launch_cvt(kv_W,    kvWbf, DH * DI,      stream);
  launch_cvt(ih_W,    ihWbf, DH * DI,      stream);
  launch_cvt(ic_W,    icWbf, DH * DI,      stream);
  launch_cvt(attn_Wh, aWhbf, DH * DH,      stream);
  launch_cvt(attn_Wk, aWkbf, DH * DH,      stream);
  launch_cvt(W_ih,    Wihbf, 4 * DH * 2 * DH, stream);
  launch_cvt(W_hh,    Whhbf, 4 * DH * DH,  stream);
  {
    int n_src = Vn * DH, n_total = Vp * DH;
    cvt_f32_bf16_pad<<<(n_total + 255) / 256, 256, 0, stream>>>(out_W, outWbf, n_src, n_total);
  }
  {
    int total = Bn * Tn * DH;
    gather_embed<<<(total + 255) / 256, 256, 0, stream>>>(emb, tgt_ids, xseqbf, total);
  }

  // --- prelude GEMMs ---
  // keys = patches @ kv_W^T + kv_b   [6272,512]
  launch_gemm(pbf, DI, kvWbf, DI, keys, DH, kv_b, Bn * Nn, DH, DI, 0, stream);
  launch_cvt(keys, keysbf, Bn * Nn * DH, stream);
  // Wk_keys = keys @ attn_Wk^T
  launch_gemm(keysbf, DH, aWkbf, DH, Wkkeys, DH, nullptr, Bn * Nn, DH, DH, 0, stream);
  // h0, c0
  launch_gemm(clsbf, DI, ihWbf, DI, h, DH, ih_b, Bn, DH, DI, 0, stream);
  launch_gemm(clsbf, DI, icWbf, DI, c, DH, ic_b, Bn, DH, DI, 0, stream);
  launch_cvt(h, hbf, Bn * DH, stream);

  // --- sequential decode ---
  dim3 lcGrid((Vn + 255) / 256, Bn);
  for (int t = 0; t < Tn; ++t) {
    // hWh = h @ attn_Wh^T
    launch_gemm(hbf, DH, aWhbf, DH, hWh, DH, nullptr, Bn, DH, DH, 0, stream);
    attn_scores<<<Bn, 256, 0, stream>>>(hWh, Wkkeys, attn_v, scores);
    softmax_ctx<<<Bn, 256, 0, stream>>>(scores, keys, xseqbf, t, zbf);
    // gates = z @ W_ih^T + h @ W_hh^T  (biases added in cell kernel)
    launch_gemm(zbf, 2 * DH, Wihbf, 2 * DH, gates, 4 * DH, nullptr, Bn, 4 * DH, 2 * DH, 0, stream);
    launch_gemm(hbf, DH,     Whhbf, DH,     gates, 4 * DH, nullptr, Bn, 4 * DH, DH,     1, stream);
    lstm_cell_ln<<<Bn, 256, 0, stream>>>(gates, b_ih, b_hh, c, ln_w, ln_b, h, hbf);
    // logits_t = h_new @ out_W^T (padded N), then +out_b into out[:, t, :]
    launch_gemm(hbf, DH, outWbf, DH, lpad, Vp, nullptr, Bn, Vp, DH, 0, stream);
    logits_bias_copy<<<lcGrid, 256, 0, stream>>>(lpad, out_b, out, t);
  }
}
